// GNN_25245817766094
// MI455X (gfx1250) — matrix operand order repaired and checked by reading, not compiled, not run on hardware
//
#include <hip/hip_runtime.h>
#include <hip/hip_bf16.h>

typedef __attribute__((ext_vector_type(16))) __bf16 v16bf;
typedef __attribute__((ext_vector_type(2)))  __bf16 v2bf;
typedef __attribute__((ext_vector_type(8)))  float  v8f;

#define IN_DIM 128
#define HID 256

// ---------------------------------------------------------------------------
// Degree / normalization kernels
// ---------------------------------------------------------------------------
__global__ void init_deg_kernel(float* __restrict__ deg, int n) {
    int i = blockIdx.x * blockDim.x + threadIdx.x;
    if (i < n) deg[i] = 1.0f;                 // self-loop contributes 1
}

__global__ void deg_edges_kernel(const int* __restrict__ dst, float* __restrict__ deg, int e) {
    int i = blockIdx.x * blockDim.x + threadIdx.x;
    if (i < e) atomicAdd(&deg[dst[i]], 1.0f);
}

__global__ void rsqrt_kernel(float* __restrict__ deg, int n) {
    int i = blockIdx.x * blockDim.x + threadIdx.x;
    if (i < n) deg[i] = rsqrtf(deg[i]);       // deg >= 1 always (self loops)
}

__global__ void zero_pooled_kernel(float* __restrict__ pooled) {
    pooled[threadIdx.x] = 0.0f;
}

// ---------------------------------------------------------------------------
// bf16 WMMA GEMM:  out[M,N] = act(A[M,K] (+bias, relu)) @ W[K,N]
// block = 256 threads = 8 waves; block tile = 64 rows x N cols
// wave (w>>1) -> row tile (16 rows), (w&1) -> column half (N/2 cols)
//
// LDS tiles are stored PRE-PERMUTED into the wave32 WMMA fragment layout:
// each lane's 16-bf16 fragment is 32 contiguous, 16B-aligned bytes
// (rows padded 16 -> 24 elements: 48B stride = 4*odd words, keeps 16B
//  alignment and spreads 16-lane b128 reads over all 64 LDS banks).
// Full-tile blocks (all but the last) take a guard-free uniform fast path.
// ---------------------------------------------------------------------------
template <int N, int K, bool BIAS_RELU>
__global__ __launch_bounds__(256) void gemm_bf16_wmma(
        const float* __restrict__ A, const float* __restrict__ W,
        const float* __restrict__ bias, float* __restrict__ out, int M) {
    constexpr int BM  = 64;
    constexpr int KT  = 32;
    constexpr int NT  = N / 32;               // 16-col tiles per wave
    constexpr int PAD = 24;                   // padded fragment-row stride (elems)

    // A: [row 0..63][hh 0..1][e 0..15 (pad 24)]
    __shared__ __align__(16) __bf16 sAp[BM * 2 * PAD];            // 6 KB
    // W: [tile 0..N/16-1][lane 0..31][e 0..15 (pad 24)]
    __shared__ __align__(16) __bf16 sWp[(N / 16) * 32 * PAD];     // N=256 -> 24 KB

    const int tid  = threadIdx.x;
    const int wave = tid >> 5;
    const int lane = tid & 31;
    const int rt   = wave >> 1;               // row tile 0..3
    const int ch   = wave & 1;                // column half
    const int r16  = lane & 15;
    const int hh   = lane >> 4;
    const int m0   = blockIdx.x * BM;
    const bool fullTile = (m0 + BM) <= M;     // block-uniform

    v8f acc[NT];
#pragma unroll
    for (int t = 0; t < NT; ++t) acc[t] = (v8f){0.f,0.f,0.f,0.f,0.f,0.f,0.f,0.f};

    for (int k0 = 0; k0 < K; k0 += KT) {
        // ---- stage A tile: f32 -> bf16, permuted to fragment layout.
        //      A-frag element e of lane(r16,hh) holds K=(e&7)+8*hh+16*(e>=8);
        //      inverse: q=k>>3 -> hh=q&1, e=(k&7)+8*(q>>1).
        if (fullTile) {
            for (int i = tid; i < BM * KT / 2; i += 256) {
                int kp = i & 15;              // KT/2 = 16 pairs per row
                int r  = i >> 4;
                int k  = kp * 2;
                const float2 p = *(const float2*)&A[(size_t)(m0 + r) * K + k0 + k];
                float a0 = p.x, a1 = p.y;
                if (BIAS_RELU) {
                    a0 += bias[k0 + k];     a0 = a0 > 0.f ? a0 : 0.f;
                    a1 += bias[k0 + k + 1]; a1 = a1 > 0.f ? a1 : 0.f;
                }
                int q = k >> 3;
                int ahh = q & 1;
                int e   = (k & 7) + 8 * (q >> 1);
                v2bf pr; pr.x = (__bf16)a0; pr.y = (__bf16)a1;
                *(v2bf*)&sAp[(r * 2 + ahh) * PAD + e] = pr;
            }
        } else {
            for (int i = tid; i < BM * KT / 2; i += 256) {
                int kp = i & 15;
                int r  = i >> 4;
                int k  = kp * 2;
                int gr = m0 + r;
                float a0 = 0.f, a1 = 0.f;
                if (gr < M) {
                    const float2 p = *(const float2*)&A[(size_t)gr * K + k0 + k];
                    a0 = p.x; a1 = p.y;
                    if (BIAS_RELU) {
                        a0 += bias[k0 + k];     a0 = a0 > 0.f ? a0 : 0.f;
                        a1 += bias[k0 + k + 1]; a1 = a1 > 0.f ? a1 : 0.f;
                    }
                }
                int q = k >> 3;
                int ahh = q & 1;
                int e   = (k & 7) + 8 * (q >> 1);
                v2bf pr; pr.x = (__bf16)a0; pr.y = (__bf16)a1;
                *(v2bf*)&sAp[(r * 2 + ahh) * PAD + e] = pr;
            }
        }
        // ---- stage W tile: f32 -> bf16, permuted to fragment layout.
        //      B-frag element e of lane L=hh*16+c holds K=(e&7)+8*hh+16*(e>=8), N=c.
        for (int i = tid; i < KT * N / 2; i += 256) {
            int n  = i % N;                   // coalesced across threads
            int kp = i / N;
            int k  = kp * 2;
            float w0 = W[(size_t)(k0 + k) * N + n];
            float w1 = W[(size_t)(k0 + k + 1) * N + n];
            int t   = n >> 4;
            int c   = n & 15;
            int whh = (k >> 3) & 1;
            int e   = (k & 7) + 8 * (k >> 4);
            v2bf pr; pr.x = (__bf16)w0; pr.y = (__bf16)w1;
            *(v2bf*)&sWp[(t * 32 + whh * 16 + c) * PAD + e] = pr;
        }
        __syncthreads();

        // ---- fragments are single 32-byte contiguous lane loads (2x b128)
        v16bf af = *(const v16bf*)&sAp[((rt * 16 + r16) * 2 + hh) * PAD];

#pragma unroll
        for (int t = 0; t < NT; ++t) {
            const int gt = ch * (N / 32) + t;          // global 16-col tile id
            v16bf bfm = *(const v16bf*)&sWp[(gt * 32 + hh * 16 + r16) * PAD];
            acc[t] = __builtin_amdgcn_wmma_f32_16x16x32_bf16(
                false, af, false, bfm, (short)0, acc[t], false, false);
        }
        __syncthreads();
    }

    // ---- store D (16x16 f32 layout: M = v + 8*half, N = lane%16)
    if (fullTile) {
#pragma unroll
        for (int t = 0; t < NT; ++t) {
            const int cbase = (ch * (N / 32) + t) * 16 + r16;
#pragma unroll
            for (int v = 0; v < 8; ++v) {
                int gr = m0 + rt * 16 + v + 8 * hh;
                out[(size_t)gr * N + cbase] = acc[t][v];
            }
        }
    } else {
#pragma unroll
        for (int t = 0; t < NT; ++t) {
            const int cbase = (ch * (N / 32) + t) * 16 + r16;
#pragma unroll
            for (int v = 0; v < 8; ++v) {
                int gr = m0 + rt * 16 + v + 8 * hh;
                if (gr < M) out[(size_t)gr * N + cbase] = acc[t][v];
            }
        }
    }
}

// ---------------------------------------------------------------------------
// Self-loop initializer: agg[i][f] = dis[i]^2 * h[i][f]  (full overwrite)
// ---------------------------------------------------------------------------
template <int F>
__global__ void selfloop_init_kernel(const float* __restrict__ h,
                                     const float* __restrict__ dis,
                                     float* __restrict__ agg, int n) {
    size_t i = (size_t)blockIdx.x * blockDim.x + threadIdx.x;
    size_t total = (size_t)n * F;
    if (i < total) {
        int r = (int)(i / F);
        float d = dis[r];
        agg[i] = d * d * h[i];
    }
}

// ---------------------------------------------------------------------------
// Edge scatter-aggregate: one wave32 per edge, coalesced feature strides.
// L2-resident (51 MB << 192 MB L2), f32 atomics at L2 atomic units.
// ---------------------------------------------------------------------------
template <int F>
__global__ __launch_bounds__(256) void edge_agg_kernel(
        const float* __restrict__ h, const int* __restrict__ src,
        const int* __restrict__ dst, const float* __restrict__ dis,
        float* __restrict__ agg, int e) {
    int w    = (int)((blockIdx.x * (size_t)blockDim.x + threadIdx.x) >> 5);
    int lane = threadIdx.x & 31;
    if (w >= e) return;
    int s = src[w], d = dst[w];
    float nw = dis[s] * dis[d];
    const float* hs = h + (size_t)s * F;
    float* ad = agg + (size_t)d * F;
#pragma unroll
    for (int j = lane; j < F; j += 32)
        atomicAdd(&ad[j], nw * hs[j]);
}

// ---------------------------------------------------------------------------
// Fused relu(x + b2) + mean-pool partial reduction over nodes
// ---------------------------------------------------------------------------
__global__ __launch_bounds__(256) void pool_kernel(const float* __restrict__ agg2,
                                                   const float* __restrict__ b2,
                                                   float* __restrict__ pooled, int n) {
    __shared__ float red[256];
    int f = threadIdx.x & 127;
    int g = threadIdx.x >> 7;
    float bb = b2[f];
    float s = 0.f;
    for (int r = blockIdx.x * 2 + g; r < n; r += gridDim.x * 2) {
        float v = agg2[(size_t)r * 128 + f] + bb;
        s += v > 0.f ? v : 0.f;
    }
    red[threadIdx.x] = s;
    __syncthreads();
    if (g == 0) atomicAdd(&pooled[f], red[f] + red[128 + f]);
}

// ---------------------------------------------------------------------------
// Final FC: out = (sum_f pooled[f] * Wfc[f]) / n + bfc
// ---------------------------------------------------------------------------
__global__ void final_kernel(const float* __restrict__ pooled,
                             const float* __restrict__ Wfc,
                             const float* __restrict__ bfc,
                             float* __restrict__ out, int n) {
    __shared__ float red[128];
    int f = threadIdx.x;
    red[f] = pooled[f] * Wfc[f];
    __syncthreads();
    for (int s = 64; s > 0; s >>= 1) {
        if (f < s) red[f] += red[f + s];
        __syncthreads();
    }
    if (f == 0) out[0] = red[0] / (float)n + bfc[0];
}

// ---------------------------------------------------------------------------
extern "C" void kernel_launch(void* const* d_in, const int* in_sizes, int n_in,
                              void* d_out, int out_size, void* d_ws, size_t ws_size,
                              hipStream_t stream) {
    const float* x   = (const float*)d_in[0];
    const int*   ei  = (const int*)d_in[1];
    const float* W1  = (const float*)d_in[2];
    const float* b1  = (const float*)d_in[3];
    const float* W2  = (const float*)d_in[4];
    const float* b2  = (const float*)d_in[5];
    const float* Wfc = (const float*)d_in[6];
    const float* bfc = (const float*)d_in[7];
    float* out = (float*)d_out;

    const int nN = in_sizes[0] / IN_DIM;   // 50000
    const int nE = in_sizes[1] / 2;        // 800000
    const int* src = ei;
    const int* dst = ei + nE;

    // workspace layout (floats): dis[nN] | pooled[128] | hbuf[nN*256] | aggbuf[nN*256]
    float* dis    = (float*)d_ws;
    float* pooled = dis + nN;
    float* hbuf   = pooled + 128;
    float* aggbuf = hbuf + (size_t)nN * HID;

    // 1) symmetric normalization coefficients dis = deg^{-1/2}
    init_deg_kernel<<<(nN + 255) / 256, 256, 0, stream>>>(dis, nN);
    deg_edges_kernel<<<(nE + 255) / 256, 256, 0, stream>>>(dst, dis, nE);
    rsqrt_kernel<<<(nN + 255) / 256, 256, 0, stream>>>(dis, nN);
    zero_pooled_kernel<<<1, 128, 0, stream>>>(pooled);

    // 2) layer 1 GEMM: h = x @ W1   [nN,128]x[128,256]
    gemm_bf16_wmma<HID, IN_DIM, false><<<(nN + 63) / 64, 256, 0, stream>>>(
        x, W1, nullptr, hbuf, nN);

    // 3) aggregation 1: agg1 = scatter(norm * h[src]) with self loops
    selfloop_init_kernel<HID><<<(int)(((size_t)nN * HID + 255) / 256), 256, 0, stream>>>(
        hbuf, dis, aggbuf, nN);
    edge_agg_kernel<HID><<<(int)(((size_t)nE * 32 + 255) / 256), 256, 0, stream>>>(
        hbuf, src, dst, dis, aggbuf, nE);

    // 4) layer 2 GEMM with fused relu(agg1 + b1): h2pre = relu(agg1+b1) @ W2
    gemm_bf16_wmma<128, HID, true><<<(nN + 63) / 64, 256, 0, stream>>>(
        aggbuf, W2, b1, hbuf, nN);

    // 5) aggregation 2
    selfloop_init_kernel<128><<<(int)(((size_t)nN * 128 + 255) / 256), 256, 0, stream>>>(
        hbuf, dis, aggbuf, nN);
    edge_agg_kernel<128><<<(int)(((size_t)nE * 32 + 255) / 256), 256, 0, stream>>>(
        hbuf, src, dst, dis, aggbuf, nE);

    // 6) fused relu(agg2 + b2) + mean pool, then final FC
    pool_kernel<<<256, 256, 0, stream>>>(aggbuf, b2, pooled, nN);
    final_kernel<<<1, 128, 0, stream>>>(pooled, Wfc, bfc, out, nN);
}
